// FasNetTAC_14259291422723
// MI455X (gfx1250) — matrix operand order repaired and checked by reading, not compiled, not run on hardware
//
#include <hip/hip_runtime.h>
#include <math.h>

// ---------------------------------------------------------------------------
// FasNetTAC forward for MI455X (gfx1250, wave32, WMMA).
// All dense contractions: v_wmma_f32_16x16x32_f16 (f16 in, f32 accum).
// Weights pre-converted to padded f16 so all fragment loads are unguarded
// vector loads (global_load_b128 / ds_load_b128).
// ---------------------------------------------------------------------------

typedef __attribute__((ext_vector_type(16))) _Float16 v16h;
typedef __attribute__((ext_vector_type(8)))  _Float16 v8h;
typedef __attribute__((ext_vector_type(8)))  float    v8f;

#define BATCH_   2
#define MICS_    4
#define NSAMP_   32000
#define WIN_     64
#define CTXL_    256
#define STRIDE_  32
#define KCTX_    576      // WIN + 2*CTX
#define FILT_    513
#define FILTP_   528      // padded to mult of 16
#define ENC_     64
#define SSEG_    1003     // (32000 + 2*320 - 576)/32 + 1
#define CHUNK_   50
#define HOP_     25
#define JCH_     43
#define HID_     128
#define GATE_    512
#define TACH_    384
#define NSPK_    2
#define BM_      8
#define EPS_     1e-8f
#define ROWS_U   (BM_*JCH_*CHUNK_)   // 17200 (mult of 16)
#define ROWS_BMS (BM_*SSEG_)         // 8024
#define PAD_BMS  8032                // padded rows
#define ROWS_GS  (BM_*NSPK_*SSEG_)   // 16048 (mult of 16)
#define JK_      (JCH_*CHUNK_)       // 2150
#define FEATK_   608                 // 577 padded to mult of 32

__device__ __forceinline__ float sigm_(float x){ return 1.f/(1.f+__expf(-x)); }

// ---------------------------------------------------------------------------
// Weight convert: dst[Npad][Kpad] (f16) = src[N][K], zero padded.
// ---------------------------------------------------------------------------
__global__ void k_w2h(const float* __restrict__ src, _Float16* __restrict__ dst,
                      int N, int K, int Npad, int Kpad)
{
  size_t idx = (size_t)blockIdx.x*256 + threadIdx.x;
  size_t total = (size_t)Npad*Kpad;
  if (idx >= total) return;
  int k = (int)(idx % Kpad);
  int n = (int)(idx / Kpad);
  dst[idx] = (n<N && k<K) ? (_Float16)src[(size_t)n*K + k] : (_Float16)0.f;
}

// ---------------------------------------------------------------------------
// WMMA GEMM:  C[Mpad,Npad] = act( A[Mpad,Kpad](f32) * W16[Npad,Kpad]^T + bias )
// Mpad mult 16, Npad mult 16, Kpad mult 32. No guards in the k-loop.
// act: 0=none 1=prelu(alpha[0]) 2=tanh 4=sigmoid()*existing C
// ---------------------------------------------------------------------------
__global__ void gemm_wmma(const float* __restrict__ A, int lda,
                          const _Float16* __restrict__ W16, int ldw,
                          float* __restrict__ C, int ldc,
                          int tn_cnt, int total_tiles, int Kr, int Nreal,
                          const float* __restrict__ bias,
                          int act, const float* __restrict__ alpha)
{
  const int wid = blockIdx.x * (blockDim.x >> 5) + (threadIdx.x >> 5);
  if (wid >= total_tiles) return;              // wave-uniform
  const int tm = wid / tn_cnt, tn = wid % tn_cnt;
  const int lane = threadIdx.x & 31;
  const int half = lane >> 4;
  const int lrow = lane & 15;
  const float*    Ar = A   + (size_t)(tm*16 + lrow)*lda;
  const _Float16* Wr = W16 + (size_t)(tn*16 + lrow)*ldw;
  union { v8f v; float e[8]; } acc;
  #pragma unroll
  for (int v=0; v<8; ++v) acc.e[v]=0.f;
  for (int k0=0; k0<Kr; k0+=32){
    __builtin_prefetch(Ar + k0 + 256, 0, 0);
    // A frag: lane rows = lrow; K runs [k0+8h, +8) and [k0+16+8h, +8)
    const float4 a0 = *(const float4*)(Ar + k0 + (half<<3));
    const float4 a1 = *(const float4*)(Ar + k0 + (half<<3) + 4);
    const float4 a2 = *(const float4*)(Ar + k0 + 16 + (half<<3));
    const float4 a3 = *(const float4*)(Ar + k0 + 16 + (half<<3) + 4);
    union { v16h v; _Float16 e[16]; } av;
    av.e[0]=(_Float16)a0.x; av.e[1]=(_Float16)a0.y; av.e[2]=(_Float16)a0.z; av.e[3]=(_Float16)a0.w;
    av.e[4]=(_Float16)a1.x; av.e[5]=(_Float16)a1.y; av.e[6]=(_Float16)a1.z; av.e[7]=(_Float16)a1.w;
    av.e[8]=(_Float16)a2.x; av.e[9]=(_Float16)a2.y; av.e[10]=(_Float16)a2.z; av.e[11]=(_Float16)a2.w;
    av.e[12]=(_Float16)a3.x; av.e[13]=(_Float16)a3.y; av.e[14]=(_Float16)a3.z; av.e[15]=(_Float16)a3.w;
    // B frag: lane col = lrow; K runs [k0+8h, +8) and [k0+16+8h, +8)
    union { v16h v; v8h h[2]; } bu;
    bu.h[0] = *(const v8h*)(Wr + k0 + (half<<3));
    bu.h[1] = *(const v8h*)(Wr + k0 + 16 + (half<<3));
    const v16h bv = bu.v;
    acc.v = __builtin_amdgcn_wmma_f32_16x16x32_f16(false, av.v, false, bv,
                                                   (short)0, acc.v, false, false);
  }
  const float al = (act==1 && alpha) ? alpha[0] : 0.f;
  const int c = tn*16 + lrow;
  if (c < Nreal){
    #pragma unroll
    for (int v=0; v<8; ++v){
      int r = tm*16 + v + (half<<3);
      float val = acc.e[v];
      if (bias) val += bias[c];
      if      (act==1) val = (val>=0.f) ? val : al*val;
      else if (act==2) val = tanhf(val);
      else if (act==4) val = sigm_(val) * C[(size_t)r*ldc + c];
      C[(size_t)r*ldc + c] = val;
    }
  }
}

// ---------------------------------------------------------------------------
// BiLSTM scan: 16 sequences per block; h (f16), c, z in LDS.
// Zin rows padded so block 22 of the intra scan reads zeros (no guards).
// ---------------------------------------------------------------------------
__global__ void lstm_scan(const float* __restrict__ Zin,
                          const _Float16* __restrict__ Whh16,  // [512][128] f16
                          float* __restrict__ Hout, int ldh,
                          int T, int rev)
{
  __shared__ _Float16 sh_h[16*HID_];
  __shared__ float    sh_c[16*HID_];
  __shared__ float    sh_z[16*GATE_];
  const int tid  = threadIdx.x;
  const int wave = tid >> 5, lane = tid & 31;
  const int row  = lane & 15, half = lane >> 4;
  const int seq0 = blockIdx.x * 16;
  for (int i=tid; i<16*HID_; i+=256){ sh_h[i]=(_Float16)0.f; sh_c[i]=0.f; }
  __syncthreads();
  for (int stp=0; stp<T; ++stp){
    const int t = rev ? (T-1-stp) : stp;
    #pragma unroll
    for (int nt=0; nt<4; ++nt){
      const int n0 = wave*64 + nt*16;
      const _Float16* Wr = Whh16 + (size_t)(n0 + row)*HID_;
      union { v8f v; float e[8]; } acc;
      #pragma unroll
      for (int v=0; v<8; ++v) acc.e[v]=0.f;
      #pragma unroll
      for (int k0=0; k0<HID_; k0+=32){
        union { v16h v; v8h h[2]; } av;
        av.h[0] = *(const v8h*)(&sh_h[row*HID_ + k0 + (half<<3)]);
        av.h[1] = *(const v8h*)(&sh_h[row*HID_ + k0 + 16 + (half<<3)]);
        union { v16h v; v8h h[2]; } bu;
        bu.h[0] = *(const v8h*)(Wr + k0 + (half<<3));
        bu.h[1] = *(const v8h*)(Wr + k0 + 16 + (half<<3));
        const v16h bv = bu.v;
        acc.v = __builtin_amdgcn_wmma_f32_16x16x32_f16(false, av.v, false, bv,
                                                       (short)0, acc.v, false, false);
      }
      #pragma unroll
      for (int v=0; v<8; ++v)
        sh_z[(v + (half<<3))*GATE_ + n0 + row] = acc.e[v];
    }
    __syncthreads();
    for (int i=tid; i<16*HID_; i+=256){
      int sq = i>>7, hh = i&127;
      size_t zr = ((size_t)(seq0+sq)*T + t)*GATE_;
      float zi = sh_z[sq*GATE_ + hh]          + Zin[zr + hh];
      float zf = sh_z[sq*GATE_ + HID_   + hh] + Zin[zr + HID_   + hh];
      float zg = sh_z[sq*GATE_ + 2*HID_ + hh] + Zin[zr + 2*HID_ + hh];
      float zo = sh_z[sq*GATE_ + 3*HID_ + hh] + Zin[zr + 3*HID_ + hh];
      float cc = sigm_(zf)*sh_c[i] + sigm_(zi)*tanhf(zg);
      float hv = sigm_(zo)*tanhf(cc);
      sh_c[i] = cc;
      sh_h[i] = (_Float16)hv;
      Hout[((size_t)(seq0+sq)*T + t)*ldh + hh] = hv;
    }
    __syncthreads();
  }
}

// ---------------------------------------------------------------------------
// Elementwise / normalization / feature kernels
// ---------------------------------------------------------------------------
__global__ void k_build_ctx(const float* __restrict__ x, float* __restrict__ ctx)
{
  size_t idx = (size_t)blockIdx.x*256 + threadIdx.x;
  const size_t total = (size_t)PAD_BMS*KCTX_;
  if (idx >= total) return;
  int k = (int)(idx % KCTX_);
  size_t g = idx / KCTX_;
  if (g >= ROWS_BMS){ ctx[idx] = 0.f; return; }   // zero pad rows
  int s  = (int)(g % SSEG_);
  int bm = (int)(g / SSEG_);
  int p = s*STRIDE_ + k - (CTXL_ + WIN_);
  ctx[idx] = (p>=0 && p<NSAMP_) ? x[(size_t)bm*NSAMP_ + p] : 0.f;
}

// gLN over enc channels (cols 0..63 of featA, ld=FEATK_), in place.
__global__ void gln_enc(float* __restrict__ F, const float* __restrict__ g,
                        const float* __restrict__ b)
{
  const int grp = blockIdx.x;
  __shared__ float s1[256], s2[256];
  const int N = SSEG_*ENC_;
  float sum=0.f, ss=0.f;
  for (int i=threadIdx.x; i<N; i+=256){
    int s=i>>6, c=i&63;
    float v = F[((size_t)grp*SSEG_+s)*FEATK_ + c];
    sum+=v; ss+=v*v;
  }
  s1[threadIdx.x]=sum; s2[threadIdx.x]=ss; __syncthreads();
  for (int st=128; st>0; st>>=1){
    if (threadIdx.x<st){ s1[threadIdx.x]+=s1[threadIdx.x+st]; s2[threadIdx.x]+=s2[threadIdx.x+st]; }
    __syncthreads();
  }
  float m = s1[0]/N, var = s2[0]/N - m*m;
  float inv = rsqrtf(var + EPS_);
  for (int i=threadIdx.x; i<N; i+=256){
    int s=i>>6, c=i&63;
    size_t q = ((size_t)grp*SSEG_+s)*FEATK_ + c;
    F[q] = g[c]*(F[q]-m)*inv + b[c];
  }
}

// Cosine features: block per (bm,s) row; writes featA cols 64..576.
__global__ void k_cos(const float* __restrict__ ctx, float* __restrict__ featA)
{
  const int gidx = blockIdx.x;          // bm*SSEG_ + s
  const int bm = gidx / SSEG_, s = gidx % SSEG_;
  const int b  = bm / MICS_;
  __shared__ float sref[KCTX_], stgt[WIN_], sred[256];
  const float* rp = ctx + (size_t)gidx*KCTX_;
  for (int i=threadIdx.x; i<KCTX_; i+=256) sref[i]=rp[i];
  const float* tp = ctx + ((size_t)(b*MICS_)*SSEG_ + s)*KCTX_ + CTXL_;
  if (threadIdx.x < WIN_) stgt[threadIdx.x] = tp[threadIdx.x];
  __syncthreads();
  float ts=0.f;
  if (threadIdx.x < WIN_) ts = stgt[threadIdx.x]*stgt[threadIdx.x];
  sred[threadIdx.x]=ts; __syncthreads();
  for (int st=128; st>0; st>>=1){
    if (threadIdx.x<st) sred[threadIdx.x]+=sred[threadIdx.x+st];
    __syncthreads();
  }
  const float tn = sqrtf(sred[0]) + EPS_;
  for (int f=threadIdx.x; f<FILT_; f+=256){
    float num=0.f, ss=0.f;
    #pragma unroll 4
    for (int t=0; t<WIN_; ++t){ float r=sref[f+t]; num += r*stgt[t]; ss += r*r; }
    featA[(size_t)gidx*FEATK_ + ENC_ + f] = num / ((sqrtf(ss)+EPS_)*tn);
  }
}

// Chunking: U[bm][j][k][c] = feat64 padded by CHUNK on s.
__global__ void k_build_u(const float* __restrict__ f64, float* __restrict__ U)
{
  size_t idx = (size_t)blockIdx.x*256 + threadIdx.x;
  const size_t total = (size_t)ROWS_U*ENC_;
  if (idx >= total) return;
  int c = (int)(idx & 63);
  size_t q = idx >> 6;
  int k = (int)(q % CHUNK_); q /= CHUNK_;
  int j = (int)(q % JCH_);
  int bm = (int)(q / JCH_);
  int s = j*HOP_ + k - CHUNK_;
  U[idx] = (s>=0 && s<SSEG_) ? f64[((size_t)bm*SSEG_+s)*ENC_ + c] : 0.f;
}

// Ut[bm][k][j][c] = U[bm][j][k][c]
__global__ void k_transpose_u(const float* __restrict__ U, float* __restrict__ Ut)
{
  size_t idx = (size_t)blockIdx.x*256 + threadIdx.x;
  const size_t total = (size_t)ROWS_U*ENC_;
  if (idx >= total) return;
  int c = (int)(idx & 63);
  size_t q = idx >> 6;
  int j = (int)(q % JCH_); q /= JCH_;
  int k = (int)(q % CHUNK_);
  int bm = (int)(q / CHUNK_);
  Ut[idx] = U[(((size_t)bm*JCH_ + j)*CHUNK_ + k)*ENC_ + c];
}

// OUT = X + gLN(H) per group (bm); transpose=1: H/X [k][j][c] -> OUT [j][k][c].
__global__ void gln_residual(const float* __restrict__ X, const float* __restrict__ H,
                             float* __restrict__ OUT,
                             const float* __restrict__ g, const float* __restrict__ b,
                             int inner, int C, int transpose, int Jd, int Kd)
{
  const int grp = blockIdx.x;
  const float* Hg = H + (size_t)grp*inner;
  __shared__ float s1[256], s2[256];
  float sum=0.f, ss=0.f;
  for (int i=threadIdx.x; i<inner; i+=256){ float v=Hg[i]; sum+=v; ss+=v*v; }
  s1[threadIdx.x]=sum; s2[threadIdx.x]=ss; __syncthreads();
  for (int st=128; st>0; st>>=1){
    if (threadIdx.x<st){ s1[threadIdx.x]+=s1[threadIdx.x+st]; s2[threadIdx.x]+=s2[threadIdx.x+st]; }
    __syncthreads();
  }
  float m = s1[0]/inner, var = s2[0]/inner - m*m;
  float inv = rsqrtf(var + EPS_);
  for (int i=threadIdx.x; i<inner; i+=256){
    int c = i % C;
    float nv = g[c]*(Hg[i]-m)*inv + b[c];
    float xv = X ? X[(size_t)grp*inner + i] : 0.f;
    size_t oi;
    if (transpose){
      int k = i/(Jd*C); int j = (i/C)%Jd;
      oi = (size_t)grp*inner + ((size_t)j*Kd + k)*C + c;
    } else oi = (size_t)grp*inner + i;
    OUT[oi] = xv + nv;
  }
}

// TAC mic-mean (rows padded to 4304): mean[(b,j,k)][h]
__global__ void k_tac_mean(const float* __restrict__ Hh, float* __restrict__ meanb,
                           const int* __restrict__ valid)
{
  size_t idx = (size_t)blockIdx.x*256 + threadIdx.x;
  const size_t total = (size_t)4304*TACH_;
  if (idx >= total) return;
  int h = (int)(idx % TACH_);
  size_t q = idx / TACH_;
  if (q >= (size_t)BATCH_*JK_){ meanb[idx]=0.f; return; }
  int r = (int)(q % JK_);
  int b = (int)(q / JK_);
  int vm = valid[b]; if (vm < 1) vm = 1;
  float acc = 0.f;
  for (int m=0; m<MICS_; ++m)
    if (m < vm) acc += Hh[((size_t)((b*MICS_+m)*JK_) + r)*TACH_ + h];
  meanb[idx] = acc / (float)vm;
}

// Cat[(b,m,j,k)][0:384]=Hh, [384:768]=mean2[(b,j,k)]
__global__ void k_tac_cat(const float* __restrict__ Hh, const float* __restrict__ mean2,
                          float* __restrict__ Cat)
{
  size_t idx = (size_t)blockIdx.x*256 + threadIdx.x;
  const size_t total = (size_t)ROWS_U*2*TACH_;
  if (idx >= total) return;
  int q = (int)(idx % (2*TACH_));
  size_t row = idx / (2*TACH_);
  if (q < TACH_) Cat[idx] = Hh[row*TACH_ + q];
  else {
    int b = (int)(row / (MICS_*JK_));
    int r = (int)(row % JK_);
    Cat[idx] = mean2[((size_t)b*JK_ + r)*TACH_ + (q - TACH_)];
  }
}

__global__ void k_prelu(const float* __restrict__ in, float* __restrict__ out,
                        const float* __restrict__ a, size_t n)
{
  size_t idx = (size_t)blockIdx.x*256 + threadIdx.x;
  if (idx >= n) return;
  float v = in[idx], al = a[0];
  out[idx] = (v>=0.f) ? v : al*v;
}

// Overlap-add over chunks: fold[g][s][c], g = bm*NSPK + spk
__global__ void k_fold(const float* __restrict__ V, float* __restrict__ fold)
{
  size_t idx = (size_t)blockIdx.x*256 + threadIdx.x;
  const size_t total = (size_t)ROWS_GS*ENC_;
  if (idx >= total) return;
  int c = (int)(idx & 63);
  size_t q = idx >> 6;
  int s = (int)(q % SSEG_);
  int g = (int)(q / SSEG_);
  int bm = g >> 1, spk = g & 1;
  int jlo = (s + HOP_) / HOP_; if (jlo < 0) jlo = 0;
  int jhi = (s + CHUNK_) / HOP_; if (jhi > JCH_-1) jhi = JCH_-1;
  float acc = 0.f;
  for (int j=jlo; j<=jhi; ++j){
    int k = s + CHUNK_ - HOP_*j;
    if (k>=0 && k<CHUNK_)
      acc += V[(((size_t)bm*JCH_ + j)*CHUNK_ + k)*(NSPK_*ENC_) + spk*ENC_ + c];
  }
  fold[idx] = 0.5f * acc;
}

// Beamform: bf[g][s][t] = sum_f ctx[bm][s][t+f] * filt[g][s][f] (filt ld=FILTP_)
__global__ void k_beamform(const float* __restrict__ ctx, const float* __restrict__ filt,
                           float* __restrict__ bf)
{
  const int gs = blockIdx.x;            // g*SSEG_ + s
  const int g = gs / SSEG_, s = gs % SSEG_;
  const int bm = g >> 1;
  __shared__ float sref[KCTX_], sfil[FILT_];
  const float* rp = ctx + ((size_t)bm*SSEG_ + s)*KCTX_;
  for (int i=threadIdx.x; i<KCTX_; i+=64) sref[i]=rp[i];
  for (int i=threadIdx.x; i<FILT_; i+=64) sfil[i]=filt[(size_t)gs*FILTP_ + i];
  __syncthreads();
  const int t = threadIdx.x;            // blockDim == 64
  float acc = 0.f;
  #pragma unroll 4
  for (int f=0; f<FILT_; ++f) acc += sref[t+f]*sfil[f];
  bf[(size_t)gs*WIN_ + t] = acc;
}

// Final window overlap-add + mic average -> out[b][spk][tau]
__global__ void k_final(const float* __restrict__ bf, const int* __restrict__ valid,
                        float* __restrict__ out)
{
  size_t idx = (size_t)blockIdx.x*256 + threadIdx.x;
  const size_t total = (size_t)BATCH_*NSPK_*NSAMP_;
  if (idx >= total) return;
  int tau = (int)(idx % NSAMP_);
  int spk = (int)((idx / NSAMP_) % NSPK_);
  int b   = (int)(idx / ((size_t)NSPK_*NSAMP_));
  int vm = valid[b]; if (vm < 1) vm = 1;
  const int p = tau + WIN_;
  int slo = (p - STRIDE_) / STRIDE_; if (slo < 0) slo = 0;
  int shi = p / STRIDE_; if (shi > SSEG_-1) shi = SSEG_-1;
  float acc = 0.f;
  for (int m=0; m<MICS_; ++m){
    if (m >= vm) continue;
    int g = (b*MICS_ + m)*NSPK_ + spk;
    for (int s=slo; s<=shi; ++s){
      int w = p - STRIDE_*s;
      if (w>=0 && w<WIN_) acc += bf[((size_t)g*SSEG_ + s)*WIN_ + w];
    }
  }
  out[idx] = acc / (float)vm;
}

// ---------------------------------------------------------------------------
// Host launcher
// ---------------------------------------------------------------------------
extern "C" void kernel_launch(void* const* d_in, const int* in_sizes, int n_in,
                              void* d_out, int out_size, void* d_ws, size_t ws_size,
                              hipStream_t stream)
{
  (void)in_sizes; (void)n_in; (void)out_size; (void)ws_size;
  const float* x      = (const float*)d_in[0];
  const int*   valid  = (const int*)  d_in[1];
  const float* enc_w  = (const float*)d_in[2];
  const float* enc_g  = (const float*)d_in[3];
  const float* enc_b  = (const float*)d_in[4];
  const float* bn_w   = (const float*)d_in[5];
  const float* c2d_a  = (const float*)d_in[68];
  const float* c2d_w  = (const float*)d_in[69];
  const float* c2d_b  = (const float*)d_in[70];
  const float* tanh_w = (const float*)d_in[71];
  const float* tanh_b = (const float*)d_in[72];
  const float* gate_w = (const float*)d_in[73];
  const float* gate_b = (const float*)d_in[74];

  float* w = (float*)d_ws;
  size_t off = 0;
  auto alloc = [&](size_t nf){
    off = (off + 7) & ~(size_t)7;
    float* p = w + off; off += (nf + 7) & ~(size_t)7; return p;
  };
  auto allocH = [&](size_t nh){
    off = (off + 7) & ~(size_t)7;
    _Float16* p = (_Float16*)(w + off); off += ((nh + 15)/16)*8; return p;
  };
  // f32 activation buffers (rows padded to mult of 16, K padded where noted)
  float* ctx    = alloc((size_t)PAD_BMS*KCTX_);      // padded rows zeroed by kernel
  float* featA  = alloc((size_t)PAD_BMS*FEATK_);     // K padded 577->608 (memset 0)
  float* feat64 = alloc((size_t)PAD_BMS*ENC_);
  float* U      = alloc((size_t)ROWS_U*ENC_);
  float* Ut     = alloc((size_t)ROWS_U*ENC_);
  float* V      = alloc((size_t)ROWS_U*NSPK_*ENC_);
  float* fold   = alloc((size_t)ROWS_GS*ENC_);
  float* bf     = alloc((size_t)ROWS_GS*WIN_);
  float* scrA   = alloc((size_t)ROWS_U*2*TACH_);     // Zin(17600x512) / Cat / tanh-gate
  float* scrB   = alloc((size_t)ROWS_U*TACH_);       // Hcat(17600x256) / Hh
  float* hproj  = alloc((size_t)ROWS_U*ENC_);
  float* meanb  = alloc((size_t)4304*TACH_);
  float* mean2  = alloc((size_t)4304*TACH_);
  // f16 weight buffers
  _Float16* enc16  = allocH((size_t)64*KCTX_);
  _Float16* bn16   = allocH((size_t)64*FEATK_);
  _Float16* c2d16  = allocH((size_t)128*64);
  _Float16* tanh16 = allocH((size_t)FILTP_*64);
  _Float16* gate16 = allocH((size_t)FILTP_*64);
  _Float16 *Wih16[2][4], *Whh16[2][4], *lw16[2][2], *inw16[2], *avgw16[2], *catw16[2];
  for (int l=0; l<2; ++l){
    for (int d=0; d<4; ++d){ Wih16[l][d]=allocH((size_t)GATE_*ENC_); Whh16[l][d]=allocH((size_t)GATE_*HID_); }
    lw16[l][0]=allocH((size_t)64*256); lw16[l][1]=allocH((size_t)64*256);
    inw16[l]=allocH((size_t)TACH_*ENC_); avgw16[l]=allocH((size_t)TACH_*TACH_);
    catw16[l]=allocH((size_t)64*2*TACH_);
  }

  auto ew = [&](size_t n){ return dim3((unsigned)((n + 255) / 256)); };
  auto w2h = [&](const float* src, _Float16* dst, int N, int K, int Np, int Kp){
    k_w2h<<<ew((size_t)Np*Kp), 256, 0, stream>>>(src, dst, N, K, Np, Kp);
  };
  auto gemm = [&](const float* A, int lda, const _Float16* W16, int ldw,
                  float* C, int ldc, int Mpad, int Npad, int Kpad, int Nreal,
                  const float* bias, int act, const float* alpha){
    int tiles = (Mpad/16) * (Npad/16);
    gemm_wmma<<<(tiles+7)/8, 256, 0, stream>>>(A, lda, W16, ldw, C, ldc,
                                               Npad/16, tiles, Kpad, Nreal,
                                               bias, act, alpha);
  };

  // ---- Weight conversions (f32 -> padded f16) -----------------------------
  w2h(enc_w,  enc16,  64, KCTX_, 64, KCTX_);
  w2h(bn_w,   bn16,   64, 577,   64, FEATK_);
  w2h(c2d_w,  c2d16,  128, 64,   128, 64);
  w2h(tanh_w, tanh16, FILT_, 64, FILTP_, 64);
  w2h(gate_w, gate16, FILT_, 64, FILTP_, 64);
  for (int l=0; l<2; ++l){
    const int base = 6 + l*31;
    const int wih_idx[4] = {0,3,10,13}, whh_idx[4] = {1,4,11,14};
    for (int d=0; d<4; ++d){
      w2h((const float*)d_in[base+wih_idx[d]], Wih16[l][d], GATE_, ENC_, GATE_, ENC_);
      w2h((const float*)d_in[base+whh_idx[d]], Whh16[l][d], GATE_, HID_, GATE_, HID_);
    }
    w2h((const float*)d_in[base+6],  lw16[l][0], 64, 256, 64, 256);
    w2h((const float*)d_in[base+16], lw16[l][1], 64, 256, 64, 256);
    w2h((const float*)d_in[base+20], inw16[l],  TACH_, ENC_, TACH_, ENC_);
    w2h((const float*)d_in[base+23], avgw16[l], TACH_, TACH_, TACH_, TACH_);
    w2h((const float*)d_in[base+26], catw16[l], 64, 2*TACH_, 64, 2*TACH_);
  }

  // ---- Front end ----------------------------------------------------------
  hipMemsetAsync(featA, 0, (size_t)PAD_BMS*FEATK_*sizeof(float), stream);
  k_build_ctx<<<ew((size_t)PAD_BMS*KCTX_), 256, 0, stream>>>(x, ctx);
  gemm(ctx, KCTX_, enc16, KCTX_, featA, FEATK_, PAD_BMS, 64, KCTX_, 64, nullptr, 0, nullptr);
  gln_enc<<<BM_, 256, 0, stream>>>(featA, enc_g, enc_b);
  k_cos<<<ROWS_BMS, 256, 0, stream>>>(ctx, featA);
  gemm(featA, FEATK_, bn16, FEATK_, feat64, ENC_, PAD_BMS, 64, FEATK_, 64, nullptr, 0, nullptr);
  k_build_u<<<ew((size_t)ROWS_U*ENC_), 256, 0, stream>>>(feat64, U);

  // ---- DPRNN + TAC layers -------------------------------------------------
  for (int layer = 0; layer < 2; ++layer){
    const int base = 6 + layer*31;
    const float* ib_f  = (const float*)d_in[base+2];
    const float* ib_b  = (const float*)d_in[base+5];
    const float* i_lb  = (const float*)d_in[base+7];
    const float* i_g   = (const float*)d_in[base+8];
    const float* i_bb  = (const float*)d_in[base+9];
    const float* eb_f  = (const float*)d_in[base+12];
    const float* eb_b  = (const float*)d_in[base+15];
    const float* e_lb  = (const float*)d_in[base+17];
    const float* e_g   = (const float*)d_in[base+18];
    const float* e_bb  = (const float*)d_in[base+19];
    const float* in_b  = (const float*)d_in[base+21];
    const float* in_a  = (const float*)d_in[base+22];
    const float* avg_b = (const float*)d_in[base+24];
    const float* avg_a = (const float*)d_in[base+25];
    const float* cat_b = (const float*)d_in[base+27];
    const float* cat_a = (const float*)d_in[base+28];
    const float* n_g   = (const float*)d_in[base+29];
    const float* n_b   = (const float*)d_in[base+30];

    // zero Zin pad rows (seqs 344..351 of the intra scans)
    hipMemsetAsync(scrA + (size_t)ROWS_U*GATE_, 0, (size_t)400*GATE_*sizeof(float), stream);

    // intra-chunk BiLSTM: nseq = 344 (22 blocks, padded to 352), T = 50
    gemm(U, ENC_, Wih16[layer][0], ENC_, scrA, GATE_, ROWS_U, GATE_, ENC_, GATE_, ib_f, 0, nullptr);
    lstm_scan<<<22, 256, 0, stream>>>(scrA, Whh16[layer][0], scrB,       256, CHUNK_, 0);
    gemm(U, ENC_, Wih16[layer][1], ENC_, scrA, GATE_, ROWS_U, GATE_, ENC_, GATE_, ib_b, 0, nullptr);
    lstm_scan<<<22, 256, 0, stream>>>(scrA, Whh16[layer][1], scrB + 128, 256, CHUNK_, 1);
    gemm(scrB, 256, lw16[layer][0], 256, hproj, ENC_, ROWS_U, 64, 256, 64, i_lb, 0, nullptr);
    gln_residual<<<BM_, 256, 0, stream>>>(U, hproj, U, i_g, i_bb,
                                          JK_*ENC_, ENC_, 0, JCH_, CHUNK_);

    // inter-chunk BiLSTM: nseq = 400 (25 blocks exact), T = 43
    k_transpose_u<<<ew((size_t)ROWS_U*ENC_), 256, 0, stream>>>(U, Ut);
    gemm(Ut, ENC_, Wih16[layer][2], ENC_, scrA, GATE_, ROWS_U, GATE_, ENC_, GATE_, eb_f, 0, nullptr);
    lstm_scan<<<25, 256, 0, stream>>>(scrA, Whh16[layer][2], scrB,       256, JCH_, 0);
    gemm(Ut, ENC_, Wih16[layer][3], ENC_, scrA, GATE_, ROWS_U, GATE_, ENC_, GATE_, eb_b, 0, nullptr);
    lstm_scan<<<25, 256, 0, stream>>>(scrA, Whh16[layer][3], scrB + 128, 256, JCH_, 1);
    gemm(scrB, 256, lw16[layer][1], 256, hproj, ENC_, ROWS_U, 64, 256, 64, e_lb, 0, nullptr);
    gln_residual<<<BM_, 256, 0, stream>>>(Ut, hproj, U, e_g, e_bb,
                                          JK_*ENC_, ENC_, 1, JCH_, CHUNK_);

    // TAC
    gemm(U, ENC_, inw16[layer], ENC_, scrB, TACH_, ROWS_U, TACH_, ENC_, TACH_, in_b, 1, in_a);
    k_tac_mean<<<ew((size_t)4304*TACH_), 256, 0, stream>>>(scrB, meanb, valid);
    gemm(meanb, TACH_, avgw16[layer], TACH_, mean2, TACH_, 4304, TACH_, TACH_, TACH_, avg_b, 1, avg_a);
    k_tac_cat<<<ew((size_t)ROWS_U*2*TACH_), 256, 0, stream>>>(scrB, mean2, scrA);
    gemm(scrA, 2*TACH_, catw16[layer], 2*TACH_, hproj, ENC_, ROWS_U, 64, 2*TACH_, 64, cat_b, 1, cat_a);
    gln_residual<<<BM_, 256, 0, stream>>>(U, hproj, U, n_g, n_b,
                                          JK_*ENC_, ENC_, 0, JCH_, CHUNK_);
  }

  // ---- Output head --------------------------------------------------------
  k_prelu<<<ew((size_t)ROWS_U*ENC_), 256, 0, stream>>>(U, Ut, c2d_a, (size_t)ROWS_U*ENC_);
  gemm(Ut, ENC_, c2d16, 64, V, NSPK_*ENC_, ROWS_U, 128, 64, 128, c2d_b, 0, nullptr);
  k_fold<<<ew((size_t)ROWS_GS*ENC_), 256, 0, stream>>>(V, fold);
  gemm(fold, ENC_, tanh16, 64, scrA, FILTP_, ROWS_GS, FILTP_, 64, FILT_, tanh_b, 2, nullptr);
  gemm(fold, ENC_, gate16, 64, scrA, FILTP_, ROWS_GS, FILTP_, 64, FILT_, gate_b, 4, nullptr);
  k_beamform<<<ROWS_GS, 64, 0, stream>>>(ctx, scrA, bf);
  k_final<<<ew((size_t)BATCH_*NSPK_*NSAMP_), 256, 0, stream>>>(bf, valid, (float*)d_out);
}
